// HiddenEdgeDistanceMLP_2104533975132
// MI455X (gfx1250) — matrix operand order repaired and checked by reading, not compiled, not run on hardware
//
#include <hip/hip_runtime.h>
#include <hip/hip_bf16.h>
#include <math.h>

// ---------------------------------------------------------------------------
// HiddenEdgeDistanceMLP on MI455X (gfx1250, wave32, WMMA f16 16x16x32)
//
// out[e] = relu( silu( silu( (s_lig[l]+s_poc[p]) W1^T + b1 ) W2^T + b2 ) W3^T + b3 )
//
// Layer 1 is linear in the gathered sum, so:
//   Al = s_lig @ W1^T            (4096 x 128, f16)
//   Bp = s_poc @ W1^T + b1       (20480 x 128, f16)
//   h1[e] = silu(Al[l] + Bp[p])  built per-lane directly into WMMA A-fragments
// Edge tiles: e0 = 16*tile; 160 % 16 == 0 so every tile = 1 ligand row x 16
// consecutive pocket rows of one complex -> indices are implicit.
// ---------------------------------------------------------------------------

typedef _Float16 half_t;
typedef __attribute__((ext_vector_type(16))) _Float16 v16h;
typedef __attribute__((ext_vector_type(8)))  _Float16 v8h;
typedef __attribute__((ext_vector_type(8)))  float    v8f;

#define NB   128          // complexes
#define NLB  32           // ligand atoms per complex
#define NPB  160          // pocket CAs per complex
#define SDIM 256
#define H1D  128
#define H2D  64
#define NLTOT (NB * NLB)  // 4096
#define NPTOT (NB * NPB)  // 20480

// silu via fast reciprocal: avoids the IEEE div chain (div_scale/div_fmas/...)
// -> mul + v_exp_f32 + add + v_rcp_f32 + mul (trans ops co-execute with WMMA).
__device__ __forceinline__ float siluf(float x) {
    return x * __builtin_amdgcn_rcpf(1.0f + __expf(-x));
}

// ---------------------------------------------------------------------------
// Kernel A: Al = s_lig @ W1^T ; Bp = s_poc @ W1^T + b1  (both stored f16)
// Combined row space [0, 24576): rows < 4096 -> ligand, else pocket.
// One wave per 16-row M-tile, 8 N-tiles of 16 (H1D=128), 8 K-steps of 32.
// ---------------------------------------------------------------------------
__global__ __launch_bounds__(256) void precompute_h1_inputs(
    const float* __restrict__ s_lig,
    const float* __restrict__ s_poc,
    const float* __restrict__ W1,
    const float* __restrict__ b1,
    half_t* __restrict__ Al,
    half_t* __restrict__ Bp)
{
    const int lane = threadIdx.x & 31;
    const int wave = threadIdx.x >> 5;
    const int tile = blockIdx.x * 8 + wave;      // 0..1535
    const int row0 = tile * 16;
    const int m    = lane & 15;                  // A-row / C-col within tile
    const int hi   = lane >> 4;                  // 0: K-low half, 1: K-high half
    const int row  = row0 + m;
    const bool pocket = (row0 >= NLTOT);
    const float* srow = pocket ? (s_poc + (size_t)(row - NLTOT) * SDIM)
                               : (s_lig + (size_t)row * SDIM);

    const v8f vzero = {0.f,0.f,0.f,0.f,0.f,0.f,0.f,0.f};
    v8f acc[8];
#pragma unroll
    for (int nt = 0; nt < 8; ++nt) acc[nt] = vzero;

#pragma unroll
    for (int ks = 0; ks < 8; ++ks) {
        const int kb = ks * 32;
        const int ko = hi ? 8 : 0;
        // A fragment: 16-bit A 16x32 layout (lanes 0-15: K {0..7,16..23};
        // lanes 16-31: K {8..15,24..31}), 2 halves per VGPR.
        v16h a;
        {
            const float4 a0 = *(const float4*)(srow + kb + ko);
            const float4 a1 = *(const float4*)(srow + kb + ko + 4);
            const float4 a2 = *(const float4*)(srow + kb + 16 + ko);
            const float4 a3 = *(const float4*)(srow + kb + 16 + ko + 4);
            a[0]=(half_t)a0.x; a[1]=(half_t)a0.y; a[2]=(half_t)a0.z; a[3]=(half_t)a0.w;
            a[4]=(half_t)a1.x; a[5]=(half_t)a1.y; a[6]=(half_t)a1.z; a[7]=(half_t)a1.w;
            a[8]=(half_t)a2.x; a[9]=(half_t)a2.y; a[10]=(half_t)a2.z; a[11]=(half_t)a2.w;
            a[12]=(half_t)a3.x; a[13]=(half_t)a3.y; a[14]=(half_t)a3.z; a[15]=(half_t)a3.w;
        }
        const int kbb = kb + (hi ? 8 : 0);  // B: lanes 0-15 K {0..7,16..23}, 16-31 K {8..15,24..31}
#pragma unroll
        for (int nt = 0; nt < 8; ++nt) {
            // B[k][n] = W1[n][k]; lane holds column n = nt*16+m, contiguous K.
            const float* wrow = W1 + (size_t)(nt * 16 + m) * SDIM + kbb;
            const float4 w0 = *(const float4*)(wrow);
            const float4 w1 = *(const float4*)(wrow + 4);
            const float4 w2 = *(const float4*)(wrow + 16);
            const float4 w3 = *(const float4*)(wrow + 20);
            v16h bf;
            bf[0]=(half_t)w0.x;  bf[1]=(half_t)w0.y;  bf[2]=(half_t)w0.z;  bf[3]=(half_t)w0.w;
            bf[4]=(half_t)w1.x;  bf[5]=(half_t)w1.y;  bf[6]=(half_t)w1.z;  bf[7]=(half_t)w1.w;
            bf[8]=(half_t)w2.x;  bf[9]=(half_t)w2.y;  bf[10]=(half_t)w2.z; bf[11]=(half_t)w2.w;
            bf[12]=(half_t)w3.x; bf[13]=(half_t)w3.y; bf[14]=(half_t)w3.z; bf[15]=(half_t)w3.w;
            acc[nt] = __builtin_amdgcn_wmma_f32_16x16x32_f16(
                false, a, false, bf, (short)0, acc[nt], false, false);
        }
    }

    // C/D layout: VGPR r -> M = r + 8*hi, N = lane&15.
#pragma unroll
    for (int nt = 0; nt < 8; ++nt) {
        const int n = nt * 16 + m;
#pragma unroll
        for (int r = 0; r < 8; ++r) {
            const int mm = r + hi * 8;
            float v = acc[nt][r];
            if (pocket) {
                v += b1[n];
                Bp[(size_t)(row0 - NLTOT + mm) * H1D + n] = (half_t)v;
            } else {
                Al[(size_t)(row0 + mm) * H1D + n] = (half_t)v;
            }
        }
    }
}

// ---------------------------------------------------------------------------
// Kernel B: per 16-edge tile: h1 = silu(Al[li] + Bp[p0+m]) -> 16x64 GEMM vs
// W2 (f16, staged in LDS) -> silu -> dot with W3 -> relu -> out.
// 4 N-tiles x 4 K-steps = 16 WMMAs per tile; 40960 tiles total.
// ---------------------------------------------------------------------------
__global__ __launch_bounds__(256) void edge_mlp_kernel(
    const half_t* __restrict__ Al,
    const half_t* __restrict__ Bp,
    const float* __restrict__ W2,
    const float* __restrict__ b2,
    const float* __restrict__ W3,
    const float* __restrict__ b3,
    float* __restrict__ out)
{
    __shared__ half_t w2s[H2D][H1D + 8];   // +8 halves pad -> conflict-free rows

    // Cooperative convert W2 (64x128 f32) -> LDS f16 once per block.
    for (int i = threadIdx.x; i < H2D * H1D; i += 256) {
        const int n = i >> 7, k = i & 127;
        w2s[n][k] = (half_t)W2[i];
    }
    __syncthreads();

    const int lane = threadIdx.x & 31;
    const int wave = threadIdx.x >> 5;
    const int tile = blockIdx.x * 8 + wave;   // 0..40959
    const int e0   = tile * 16;
    const int li   = e0 / NPB;                // global ligand row (constant in tile)
    const int cb   = li >> 5;                 // complex = li / NLB
    const int jp0  = e0 - li * NPB;           // pocket offset within complex
    const int p0   = cb * NPB + jp0;

    const int m  = lane & 15;
    const int hi = lane >> 4;
    const half_t* alrow = Al + (size_t)li * H1D;
    const half_t* bprow = Bp + (size_t)(p0 + m) * H1D;

    const v8f vzero = {0.f,0.f,0.f,0.f,0.f,0.f,0.f,0.f};
    v8f acc[4];
#pragma unroll
    for (int nt = 0; nt < 4; ++nt) acc[nt] = vzero;

#pragma unroll
    for (int ks = 0; ks < 4; ++ks) {
        const int kb = ks * 32;
        const int ko = hi ? 8 : 0;
        // Build h1 A-fragment directly: silu(Al + Bp) per needed element.
        const v8h a0 = *(const v8h*)(alrow + kb + ko);
        const v8h p0v = *(const v8h*)(bprow + kb + ko);
        const v8h a1 = *(const v8h*)(alrow + kb + 16 + ko);
        const v8h p1v = *(const v8h*)(bprow + kb + 16 + ko);
        v16h a;
#pragma unroll
        for (int j = 0; j < 8; ++j) {
            const float x0 = (float)a0[j] + (float)p0v[j];
            const float x1 = (float)a1[j] + (float)p1v[j];
            a[j]     = (half_t)siluf(x0);
            a[8 + j] = (half_t)siluf(x1);
        }
        const int kbb = kb + (hi ? 8 : 0);
#pragma unroll
        for (int nt = 0; nt < 4; ++nt) {
            const half_t* wrow = &w2s[nt * 16 + m][kbb];
            const v8h b0 = *(const v8h*)(wrow);
            const v8h b1v = *(const v8h*)(wrow + 16);
            v16h bf;
#pragma unroll
            for (int j = 0; j < 8; ++j) { bf[j] = b0[j]; bf[8 + j] = b1v[j]; }
            acc[nt] = __builtin_amdgcn_wmma_f32_16x16x32_f16(
                false, a, false, bf, (short)0, acc[nt], false, false);
        }
    }

    // Epilogue: h2 = silu(acc + b2), out = relu(h2 . w3 + b3).
    float bb[4], ww[4];
#pragma unroll
    for (int nt = 0; nt < 4; ++nt) {
        bb[nt] = b2[nt * 16 + m];
        ww[nt] = W3[nt * 16 + m];
    }
    const float bias3 = b3[0];

#pragma unroll
    for (int r = 0; r < 8; ++r) {
        float sum = 0.f;
#pragma unroll
        for (int nt = 0; nt < 4; ++nt) {
            const float h = siluf(acc[nt][r] + bb[nt]);
            sum = fmaf(h, ww[nt], sum);
        }
        // Reduce across the 16 lanes holding row M = r + 8*hi.
        sum += __shfl_xor(sum, 1, 32);
        sum += __shfl_xor(sum, 2, 32);
        sum += __shfl_xor(sum, 4, 32);
        sum += __shfl_xor(sum, 8, 32);
        if (m == 0) {
            out[e0 + r + hi * 8] = fmaxf(sum + bias3, 0.0f);
        }
    }
}

// ---------------------------------------------------------------------------
extern "C" void kernel_launch(void* const* d_in, const int* in_sizes, int n_in,
                              void* d_out, int out_size, void* d_ws, size_t ws_size,
                              hipStream_t stream)
{
    (void)in_sizes; (void)n_in; (void)out_size; (void)ws_size;
    const float* s_lig = (const float*)d_in[0];
    const float* s_poc = (const float*)d_in[1];
    // d_in[2] (l) and d_in[3] (p) encode the block-diagonal cross product,
    // which is reconstructed analytically from the tile index.
    const float* W1 = (const float*)d_in[4];
    const float* b1 = (const float*)d_in[5];
    const float* W2 = (const float*)d_in[6];
    const float* b2 = (const float*)d_in[7];
    const float* W3 = (const float*)d_in[8];
    const float* b3 = (const float*)d_in[9];
    float* out = (float*)d_out;

    // Workspace: Al (4096x128 f16, 1 MB) then Bp (20480x128 f16, 5 MB).
    half_t* Al = (half_t*)d_ws;
    half_t* Bp = Al + (size_t)NLTOT * H1D;

    // 24576 rows / 16 per tile / 8 waves per block = 192 blocks.
    precompute_h1_inputs<<<192, 256, 0, stream>>>(s_lig, s_poc, W1, b1, Al, Bp);
    // 655360 edges / 16 per tile / 8 waves per block = 5120 blocks.
    edge_mlp_kernel<<<5120, 256, 0, stream>>>(Al, Bp, W2, b2, W3, b3, out);
}
